// CRSDCell_71339406786971
// MI455X (gfx1250) — compile-verified, matrix operand order checked by reading, not run on hardware
//
#include <hip/hip_runtime.h>
#include <hip/hip_bf16.h>
#include <math.h>

typedef __bf16 bf16;
typedef __attribute__((ext_vector_type(16))) __bf16 v16bf;
typedef __attribute__((ext_vector_type(8)))  __bf16 v8bf;
typedef __attribute__((ext_vector_type(8)))  float  v8f;

#define NB      65536          // batch
#define TILE_M  128
#define NBLOCKS (NB / TILE_M)  // 512

// ---------------------------------------------------------------------------
// Per-wave 16x16 WMMA over one K-segment with uniform A source.
// A fragment (16-bit ISA layout): lane%16 = row (baked into `a` by caller),
// lane/16 = grp selects K-halves; elems 0..7 = K+grp*8, elems 8..15 = K+16+grp*8.
// B mirrors with lane%16 = output column (baked into `w`).
// All loads are 16-byte (global_load_b128 / ds_load_b128).
// ---------------------------------------------------------------------------
__device__ __forceinline__ v8f gemm_seg(const bf16* a, const bf16* w, int Kseg,
                                        v8f acc, int grp) {
  a += grp * 8;
  w += grp * 8;
  for (int k = 0; k < Kseg; k += 32) {
    v8bf alo = *(const v8bf*)(a + k);
    v8bf ahi = *(const v8bf*)(a + k + 16);
    v8bf blo = *(const v8bf*)(w + k);
    v8bf bhi = *(const v8bf*)(w + k + 16);
    v16bf av = __builtin_shufflevector(alo, ahi, 0,1,2,3,4,5,6,7,
                                                 8,9,10,11,12,13,14,15);
    v16bf bv = __builtin_shufflevector(blo, bhi, 0,1,2,3,4,5,6,7,
                                                 8,9,10,11,12,13,14,15);
    acc = __builtin_amdgcn_wmma_f32_16x16x32_bf16(false, av, false, bv,
                                                  (short)0, acc, false, false);
  }
  return acc;
}

// ---------------------------------------------------------------------------
// Kernel 0: zero accumulators (Macc[4096] | ksum[64] | vsum[64])
// ---------------------------------------------------------------------------
__global__ void k_zero(float* p, int n) {
  int i = blockIdx.x * blockDim.x + threadIdx.x;
  if (i < n) p[i] = 0.f;
}

// ---------------------------------------------------------------------------
// Kernel W: pack all weights to bf16, fused/concatenated layouts (one-off)
// ---------------------------------------------------------------------------
struct PW {
  const float *Wx0,*Wh0,*Wx1,*Wh1,*Wx2,*Wh2,*Aw,*Bw,*Uw,*keyw,*valw,*gatew,*rmw;
  bf16 *Wres,*Wht,*Wkey,*Wval,*Wgate,*Wrm;
};
__global__ void k_pack_w(PW p) {
  int i = blockIdx.x * blockDim.x + threadIdx.x;
  const int n0 = 512*768, n1 = 512*1280, n2 = 64*768, n3 = 64*512,
            n4 = 512*1152, n5 = 512*64;
  if (i < n0) {                       // W_res[512][768] = [Wx_i | Wh_i] stacked
    int r = i / 768, c = i % 768;
    const float *Wx, *Wh; int rl;
    if (r < 128)      { Wx = p.Wx0; Wh = p.Wh0; rl = r;       }
    else if (r < 256) { Wx = p.Wx1; Wh = p.Wh1; rl = r - 128; }
    else              { Wx = p.Wx2; Wh = p.Wh2; rl = r - 256; }
    float v = (c < 256) ? Wx[rl*256 + c] : Wh[rl*512 + (c-256)];
    p.Wres[i] = (bf16)v; return;
  }
  i -= n0;
  if (i < n1) {                       // W_ht[512][1280] = [A | B | U]
    int r = i / 1280, c = i % 1280;
    float v = (c < 512)  ? p.Aw[r*512 + c]
            : (c < 1024) ? p.Bw[r*512 + (c-512)]
                         : p.Uw[r*256 + (c-1024)];
    p.Wht[i] = (bf16)v; return;
  }
  i -= n1;
  if (i < n2) { p.Wkey[i]  = (bf16)p.keyw[i];  return; }
  i -= n2;
  if (i < n3) { p.Wval[i]  = (bf16)p.valw[i];  return; }
  i -= n3;
  if (i < n4) { p.Wgate[i] = (bf16)p.gatew[i]; return; }
  i -= n4;
  if (i < n5) { p.Wrm[i]   = (bf16)p.rmw[i]; }
}

// ---------------------------------------------------------------------------
// Kernel A: pack activations x, h_prev to bf16 (8 elems / thread, b128 I/O)
// ---------------------------------------------------------------------------
__global__ void k_pack_act(const float* __restrict__ x, const float* __restrict__ h,
                           bf16* __restrict__ xb, bf16* __restrict__ hb) {
  size_t i = (size_t)blockIdx.x * blockDim.x + threadIdx.x;
  const size_t nx = (size_t)NB * 256 / 8;
  const size_t nh = (size_t)NB * 512 / 8;
  const float* s; bf16* d;
  if (i < nx)            { s = x + i * 8;        d = xb + i * 8; }
  else if (i < nx + nh)  { size_t j = i - nx; s = h + j * 8; d = hb + j * 8; }
  else return;
  float4 a = *(const float4*)s;
  float4 b = *(const float4*)(s + 4);
  v8bf o;
  o[0]=(bf16)a.x; o[1]=(bf16)a.y; o[2]=(bf16)a.z; o[3]=(bf16)a.w;
  o[4]=(bf16)b.x; o[5]=(bf16)b.y; o[6]=(bf16)b.z; o[7]=(bf16)b.w;
  *(v8bf*)d = o;
}

// ---------------------------------------------------------------------------
// Kernel 1: reservoirs + h_tilde + k + v + batch reductions (128-row tiles)
// dyn LDS: s_rcat[128][512]bf16 (128KB) | s_k[128][64]bf16 | s_v[128][64]bf16
// ---------------------------------------------------------------------------
struct P1 {
  const bf16 *x_bf, *h_bf;                       // packed activations
  const float *r0p, *r1p, *r2p, *alpha;
  const float *bx0,*bh0,*bx1,*bh1,*bx2,*bh2,*Ub,*keyb,*valb;
  const bf16 *Wres,*Wht,*Wkey,*Wval;
  float *outR0,*outR1,*outR2;                    // reservoir outputs (f32)
  bf16  *rcat_bf, *ht_bf;                        // bf16 copies for later GEMMs
  float *ws_ht, *ws_k;                           // f32 for final mix / normalize
  float *Macc, *ksum, *vsum;                     // cross-batch accumulators
};
__global__ __launch_bounds__(256) void k_cell_fwd(P1 p) {
  extern __shared__ char smem[];
  bf16* s_rcat = (bf16*)smem;                    // [128][512]
  bf16* s_k    = (bf16*)(smem + 128*512*2);      // [128][64]
  bf16* s_v    = s_k + 128*64;                   // [128][64]

  const int r0   = blockIdx.x * TILE_M;
  const int wv   = threadIdx.x >> 5;
  const int lane = threadIdx.x & 31;
  const int idx  = lane & 15, grp = lane >> 4;
  const v8f vzero = {};

  // ---- Phase 1: reservoirs: pre = [x|h] @ Wres^T ; r' = a*r + (1-a)*tanh ----
  for (int t = wv; t < 256; t += 8) {
    int rb = t >> 5, cb = t & 31;
    int rbase = r0 + rb * 16;
    const bf16* ax = p.x_bf + (size_t)(rbase + idx) * 256;
    const bf16* ah = p.h_bf + (size_t)(rbase + idx) * 512;
    const bf16* wr = p.Wres + (size_t)(cb*16 + idx) * 768;
    v8f acc = gemm_seg(ax, wr,       256, vzero, grp);
    acc     = gemm_seg(ah, wr + 256, 512, acc,   grp);

    int col = cb*16 + idx;
    const float *rprev, *bx, *bh; float* outp; int w, cl, ai;
    if (col < 128)      { ai=0; cl=col;     w=128; rprev=p.r0p; bx=p.bx0; bh=p.bh0; outp=p.outR0; }
    else if (col < 256) { ai=1; cl=col-128; w=128; rprev=p.r1p; bx=p.bx1; bh=p.bh1; outp=p.outR1; }
    else                { ai=2; cl=col-256; w=256; rprev=p.r2p; bx=p.bx2; bh=p.bh2; outp=p.outR2; }
    float a    = 1.f / (1.f + __expf(-p.alpha[ai]));
    float bias = bx[cl] + bh[cl];
    #pragma unroll
    for (int j = 0; j < 8; ++j) {
      int rt = rb*16 + j + grp*8;
      size_t r = (size_t)(r0 + rt);
      float pre = acc[j] + bias;
      float val = a * rprev[r*w + cl] + (1.f - a) * tanhf(pre);
      outp[r*w + cl] = val;
      bf16 vb = (bf16)val;
      s_rcat[rt*512 + col] = vb;
      p.rcat_bf[r*512 + col] = vb;
    }
  }
  __syncthreads();

  // ---- Phase 2: h_tilde = gelu([h | r_cat | x] @ Wht^T + Ub) ----
  for (int t = wv; t < 256; t += 8) {
    int rb = t >> 5, cb = t & 31;
    int rbase = r0 + rb*16, rtb = rb*16;
    const bf16* wr = p.Wht + (size_t)(cb*16 + idx) * 1280;
    v8f acc = gemm_seg(p.h_bf + (size_t)(rbase+idx)*512, wr,        512, vzero, grp);
    acc     = gemm_seg(s_rcat + (rtb+idx)*512,           wr + 512,  512, acc,   grp);
    acc     = gemm_seg(p.x_bf + (size_t)(rbase+idx)*256, wr + 1024, 256, acc,   grp);

    int col = cb*16 + idx;
    float bias = p.Ub[col];
    #pragma unroll
    for (int j = 0; j < 8; ++j) {
      size_t r = (size_t)(r0 + rb*16 + j + grp*8);
      float pre = acc[j] + bias;
      float ht  = 0.5f * pre * (1.f + erff(pre * 0.70710678f)); // exact gelu
      p.ws_ht[r*512 + col] = ht;
      p.ht_bf[r*512 + col] = (bf16)ht;
    }
  }
  __threadfence_block();   // make ht_bf visible to other waves in this block
  __syncthreads();

  // ---- Phase 3: k = [r_cat | x] @ Wkey^T + keyb ----
  for (int t = wv; t < 32; t += 8) {
    int rb = t >> 2, cb = t & 3;
    int rtb = rb*16, rbase = r0 + rtb;
    const bf16* wr = p.Wkey + (size_t)(cb*16 + idx) * 768;
    v8f acc = gemm_seg(s_rcat + (rtb+idx)*512,           wr,       512, vzero, grp);
    acc     = gemm_seg(p.x_bf + (size_t)(rbase+idx)*256, wr + 512, 256, acc,   grp);
    int col = cb*16 + idx;
    float bias = p.keyb[col];
    #pragma unroll
    for (int j = 0; j < 8; ++j) {
      int rt = rtb + j + grp*8;
      float kv = acc[j] + bias;
      p.ws_k[(size_t)(r0+rt)*64 + col] = kv;
      s_k[rt*64 + col] = (bf16)kv;
    }
  }

  // ---- Phase 4: v = h_tilde @ Wval^T + valb ----
  for (int t = wv; t < 32; t += 8) {
    int rb = t >> 2, cb = t & 3;
    int rtb = rb*16, rbase = r0 + rtb;
    const bf16* wr = p.Wval + (size_t)(cb*16 + idx) * 512;
    v8f acc = gemm_seg(p.ht_bf + (size_t)(rbase+idx)*512, wr, 512, vzero, grp);
    int col = cb*16 + idx;
    float bias = p.valb[col];
    #pragma unroll
    for (int j = 0; j < 8; ++j) {
      int rt = rtb + j + grp*8;
      s_v[rt*64 + col] = (bf16)(acc[j] + bias);
    }
  }
  __syncthreads();

  // ---- Phase 5: batch reductions: sum(k), sum(v), partial k^T v ----
  int tid = threadIdx.x;
  if (tid < 64) {
    float s = 0.f;
    for (int r = 0; r < 128; ++r) s += (float)s_k[r*64 + tid];
    atomicAdd(&p.ksum[tid], s);
  } else if (tid < 128) {
    int c = tid - 64; float s = 0.f;
    for (int r = 0; r < 128; ++r) s += (float)s_v[r*64 + c];
    atomicAdd(&p.vsum[c], s);
  }
  {
    int dk = tid >> 2, db = (tid & 3) * 16;
    float acc[16];
    #pragma unroll
    for (int j = 0; j < 16; ++j) acc[j] = 0.f;
    for (int r = 0; r < 128; ++r) {
      float kv = (float)s_k[r*64 + dk];
      #pragma unroll
      for (int j = 0; j < 16; ++j) acc[j] += kv * (float)s_v[r*64 + db + j];
    }
    #pragma unroll
    for (int j = 0; j < 16; ++j) atomicAdd(&p.Macc[dk*64 + db + j], acc[j]);
  }
}

// ---------------------------------------------------------------------------
// Kernel 2: finalize K (slots x dk), V^T (dv x slots), M^T (dv x dk) in bf16
// ---------------------------------------------------------------------------
struct P2 {
  const float *buf_keys, *buf_vals, *hebbM;
  const float *ksum, *vsum, *Macc;
  bf16 *Kbf, *Vt, *Mt;
};
__global__ void k_finalize(P2 p) {
  const float invB = 1.f / (float)NB;
  for (int t = threadIdx.x; t < 4096; t += blockDim.x) {
    int a = t >> 6, b = t & 63;
    float kv = (a == 0) ? p.ksum[b] * invB : p.buf_keys[a*64 + b];
    p.Kbf[t] = (bf16)kv;                                    // K[s][dk]
    float vv = (b == 0) ? p.vsum[a] * invB : p.buf_vals[b*64 + a];
    p.Vt[t] = (bf16)vv;                                     // V^T[dv][s]
    float mv = 0.9f * p.hebbM[b*64 + a] + 0.1f * p.Macc[b*64 + a] * invB;
    p.Mt[t] = (bf16)mv;                                     // M^T[dv][dk]
  }
}

// ---------------------------------------------------------------------------
// Kernel 3: c = normalize(k); attn softmax; v_hat; gate; chi; h_new
// ---------------------------------------------------------------------------
struct P3 {
  const bf16 *h_bf, *rcat_bf;
  const float *gateb, *rmb;
  const float *ws_k, *ws_ht;
  const bf16 *Kbf, *Vt, *Mt, *Wgate, *Wrm;
  float *h_new;
};
__global__ __launch_bounds__(256) void k_gate(P3 p) {
  __shared__ __align__(16) bf16 s_c[128*64];
  __shared__ __align__(16) bf16 s_attn[128*64];
  __shared__ __align__(16) bf16 s_vhat[128*64];
  const int r0   = blockIdx.x * TILE_M;
  const int wv   = threadIdx.x >> 5;
  const int lane = threadIdx.x & 31;
  const int idx  = lane & 15, grp = lane >> 4;
  const v8f vzero = {};

  // ---- c = k / max(||k||, eps) ----
  if (threadIdx.x < 128) {
    int rt = threadIdx.x;
    const float* kp = p.ws_k + (size_t)(r0 + rt) * 64;
    float ss = 0.f;
    for (int i = 0; i < 64; ++i) { float kv = kp[i]; ss += kv * kv; }
    float inv = 1.f / fmaxf(sqrtf(ss), 1e-12f);
    for (int i = 0; i < 64; ++i) s_c[rt*64 + i] = (bf16)(kp[i] * inv);
  }
  __syncthreads();

  // ---- attn = softmax(c @ K^T / 8): each wave owns 16 rows, all 64 slots ----
  {
    int rtb = wv * 16;
    const bf16* ac = s_c + (rtb + idx) * 64;
    v8f l0 = gemm_seg(ac, p.Kbf + ( 0 + idx)*64, 64, vzero, grp);
    v8f l1 = gemm_seg(ac, p.Kbf + (16 + idx)*64, 64, vzero, grp);
    v8f l2 = gemm_seg(ac, p.Kbf + (32 + idx)*64, 64, vzero, grp);
    v8f l3 = gemm_seg(ac, p.Kbf + (48 + idx)*64, 64, vzero, grp);
    #pragma unroll
    for (int j = 0; j < 8; ++j) {
      float a0 = l0[j]*0.125f, a1 = l1[j]*0.125f, a2 = l2[j]*0.125f, a3 = l3[j]*0.125f;
      float m = fmaxf(fmaxf(a0, a1), fmaxf(a2, a3));
      #pragma unroll
      for (int msk = 1; msk < 16; msk <<= 1) m = fmaxf(m, __shfl_xor(m, msk, 32));
      float e0 = __expf(a0 - m), e1 = __expf(a1 - m), e2 = __expf(a2 - m), e3 = __expf(a3 - m);
      float s = e0 + e1 + e2 + e3;
      #pragma unroll
      for (int msk = 1; msk < 16; msk <<= 1) s += __shfl_xor(s, msk, 32);
      float inv = 1.f / s;
      int rt = rtb + j + grp*8;
      s_attn[rt*64 +      idx] = (bf16)(e0 * inv);
      s_attn[rt*64 + 16 + idx] = (bf16)(e1 * inv);
      s_attn[rt*64 + 32 + idx] = (bf16)(e2 * inv);
      s_attn[rt*64 + 48 + idx] = (bf16)(e3 * inv);
    }
  }
  __syncthreads();

  // ---- v_hat = 0.5*(attn @ V + c @ M) ----
  {
    int rtb = wv * 16;
    const bf16* aa = s_attn + (rtb + idx) * 64;
    const bf16* ac = s_c    + (rtb + idx) * 64;
    #pragma unroll
    for (int cb = 0; cb < 4; ++cb) {
      v8f ab = gemm_seg(aa, p.Vt + (cb*16 + idx)*64, 64, vzero, grp);
      v8f hb = gemm_seg(ac, p.Mt + (cb*16 + idx)*64, 64, vzero, grp);
      int col = cb*16 + idx;
      #pragma unroll
      for (int j = 0; j < 8; ++j) {
        int rt = rtb + j + grp*8;
        s_vhat[rt*64 + col] = (bf16)(0.5f * (ab[j] + hb[j]));
      }
    }
  }
  __syncthreads();

  // ---- g = sigmoid([h|r_cat|c|v_hat] @ Wgate^T); chi = v_hat @ Wrm^T; mix ----
  for (int t = wv; t < 256; t += 8) {
    int rb = t >> 5, cb = t & 31;
    int rtb = rb*16; size_t rbase = (size_t)(r0 + rtb);
    const bf16* wr = p.Wgate + (size_t)(cb*16 + idx) * 1152;
    const bf16* avh = s_vhat + (rtb + idx) * 64;
    v8f g = gemm_seg(p.h_bf    + (rbase + idx)*512, wr,        512, vzero, grp);
    g     = gemm_seg(p.rcat_bf + (rbase + idx)*512, wr + 512,  512, g,     grp);
    g     = gemm_seg(s_c       + (rtb   + idx)*64,  wr + 1024,  64, g,     grp);
    g     = gemm_seg(avh,                           wr + 1088,  64, g,     grp);
    v8f ch = gemm_seg(avh, p.Wrm + (size_t)(cb*16 + idx)*64, 64, vzero, grp);

    int col = cb*16 + idx;
    float gb = p.gateb[col], rb2 = p.rmb[col];
    #pragma unroll
    for (int j = 0; j < 8; ++j) {
      size_t r = rbase + j + grp*8;
      float gg  = 1.f / (1.f + __expf(-(g[j] + gb)));
      float chi = ch[j] + rb2;
      float ht  = p.ws_ht[r*512 + col];
      p.h_new[r*512 + col] = (1.f - gg) * ht + gg * chi;
    }
  }
}

// ---------------------------------------------------------------------------
extern "C" void kernel_launch(void* const* d_in, const int* in_sizes, int n_in,
                              void* d_out, int out_size, void* d_ws, size_t ws_size,
                              hipStream_t stream) {
  const float* x      = (const float*)d_in[0];
  const float* h_prev = (const float*)d_in[1];
  const float* r0p    = (const float*)d_in[2];
  const float* r1p    = (const float*)d_in[3];
  const float* r2p    = (const float*)d_in[4];
  const float* bkeys  = (const float*)d_in[5];
  const float* bvals  = (const float*)d_in[6];
  const float* hebbM  = (const float*)d_in[7];
  const float* alpha  = (const float*)d_in[8];
  const float* Wx0 = (const float*)d_in[9],  *bx0 = (const float*)d_in[10];
  const float* Wh0 = (const float*)d_in[11], *bh0 = (const float*)d_in[12];
  const float* Wx1 = (const float*)d_in[13], *bx1 = (const float*)d_in[14];
  const float* Wh1 = (const float*)d_in[15], *bh1 = (const float*)d_in[16];
  const float* Wx2 = (const float*)d_in[17], *bx2 = (const float*)d_in[18];
  const float* Wh2 = (const float*)d_in[19], *bh2 = (const float*)d_in[20];
  const float* Aw  = (const float*)d_in[21];
  const float* Bw  = (const float*)d_in[22];
  const float* Uw  = (const float*)d_in[23], *Ub   = (const float*)d_in[24];
  const float* keyw = (const float*)d_in[25], *keyb = (const float*)d_in[26];
  const float* valw = (const float*)d_in[27], *valb = (const float*)d_in[28];
  const float* gatew = (const float*)d_in[29], *gateb = (const float*)d_in[30];
  const float* rmw   = (const float*)d_in[31], *rmb   = (const float*)d_in[32];

  float* out  = (float*)d_out;
  float* hnew = out;
  float* outR0 = out + (size_t)NB * 512;
  float* outR1 = outR0 + (size_t)NB * 128;
  float* outR2 = outR1 + (size_t)NB * 128;

  // ---- workspace carve-up ----
  char* ws = (char*)d_ws;
  size_t off = 0;
  auto carve = [&](size_t bytes) -> char* {
    char* pp = ws + off;
    off += (bytes + 255) & ~(size_t)255;
    return pp;
  };
  bf16* Wres  = (bf16*)carve(512*768  * sizeof(bf16));
  bf16* Wht   = (bf16*)carve(512*1280 * sizeof(bf16));
  bf16* Wkey  = (bf16*)carve(64*768   * sizeof(bf16));
  bf16* Wval  = (bf16*)carve(64*512   * sizeof(bf16));
  bf16* Wgate = (bf16*)carve(512*1152 * sizeof(bf16));
  bf16* Wrm   = (bf16*)carve(512*64   * sizeof(bf16));
  bf16* Kbf   = (bf16*)carve(4096 * sizeof(bf16));
  bf16* Vt    = (bf16*)carve(4096 * sizeof(bf16));
  bf16* Mt    = (bf16*)carve(4096 * sizeof(bf16));
  float* accs = (float*)carve(4224 * sizeof(float));     // Macc|ksum|vsum
  float* Macc = accs, *ksum = accs + 4096, *vsum = accs + 4160;
  float* ws_k   = (float*)carve((size_t)NB * 64  * sizeof(float));
  float* ws_ht  = (float*)carve((size_t)NB * 512 * sizeof(float));
  bf16*  x_bf   = (bf16*)carve((size_t)NB * 256 * sizeof(bf16));
  bf16*  h_bf   = (bf16*)carve((size_t)NB * 512 * sizeof(bf16));
  bf16*  rcat_bf= (bf16*)carve((size_t)NB * 512 * sizeof(bf16));
  bf16*  ht_bf  = (bf16*)carve((size_t)NB * 512 * sizeof(bf16));
  (void)ws_size; (void)in_sizes; (void)n_in; (void)out_size;

  // ---- 0: zero accumulators (ws is poisoned; must be deterministic) ----
  k_zero<<<(4224 + 255) / 256, 256, 0, stream>>>(accs, 4224);

  // ---- W: pack weights to bf16 ----
  PW pw = { Wx0,Wh0,Wx1,Wh1,Wx2,Wh2,Aw,Bw,Uw,keyw,valw,gatew,rmw,
            Wres,Wht,Wkey,Wval,Wgate,Wrm };
  const int npack = 512*768 + 512*1280 + 64*768 + 64*512 + 512*1152 + 512*64;
  k_pack_w<<<(npack + 255) / 256, 256, 0, stream>>>(pw);

  // ---- A: pack activations to bf16 ----
  const size_t nact8 = ((size_t)NB * 256 + (size_t)NB * 512) / 8;
  k_pack_act<<<(unsigned)((nact8 + 255) / 256), 256, 0, stream>>>(x, h_prev, x_bf, h_bf);

  // ---- 1: main cell GEMMs + reductions ----
  P1 p1 = { x_bf, h_bf, r0p, r1p, r2p, alpha,
            bx0, bh0, bx1, bh1, bx2, bh2, Ub, keyb, valb,
            Wres, Wht, Wkey, Wval,
            outR0, outR1, outR2,
            rcat_bf, ht_bf,
            ws_ht, ws_k,
            Macc, ksum, vsum };
  size_t shmem1 = (size_t)(128*512 + 128*64 + 128*64) * sizeof(bf16); // 160 KB
  k_cell_fwd<<<NBLOCKS, 256, shmem1, stream>>>(p1);

  // ---- 2: finalize episodic buffer + Hebbian matrix ----
  P2 p2 = { bkeys, bvals, hebbM, ksum, vsum, Macc, Kbf, Vt, Mt };
  k_finalize<<<1, 256, 0, stream>>>(p2);

  // ---- 3: attention + gate + output ----
  P3 p3 = { h_bf, rcat_bf, gateb, rmb, ws_k, ws_ht,
            Kbf, Vt, Mt, Wgate, Wrm, hnew };
  k_gate<<<NBLOCKS, 256, 0, stream>>>(p3);
}
